// DeepFM_31387620999258
// MI455X (gfx1250) — matrix operand order repaired and checked by reading, not compiled, not run on hardware
//
#include <hip/hip_runtime.h>

#define B_    16384
#define NFEAT 32
#define D_    64
#define K1    2048   // NFEAT*D_
#define H1    1024
#define H2    512
#define HOF_  64
#define NP    496    // 32*31/2
#define EPS_  1e-5f

typedef __attribute__((ext_vector_type(8)))  __bf16 v8bf;
typedef __attribute__((ext_vector_type(16))) __bf16 v16bf;
typedef __attribute__((ext_vector_type(8)))  float  v8f;

static __device__ inline v8bf load8(const __bf16* p) {
    return *(const v8bf*)p;
}
static __device__ inline v16bf cat16(v8bf lo, v8bf hi) {
    return __builtin_shufflevector(lo, hi, 0,1,2,3,4,5,6,7,8,9,10,11,12,13,14,15);
}

// ---- CDNA5 async global->LDS copy (ASYNCcnt-tracked), per-lane 16 bytes ----
static __device__ inline void async_copy_b128(unsigned lds_off, const void* gaddr) {
    asm volatile("global_load_async_to_lds_b128 %0, %1, off"
                 :: "v"(lds_off), "v"(gaddr)
                 : "memory");
}
static __device__ inline void wait_async0() {
    asm volatile("s_wait_asynccnt 0" ::: "memory");
}

// ---------------- conversion / packing ----------------

__global__ __launch_bounds__(256) void k_cvt_bf16(const float* __restrict__ src,
                                                  __bf16* __restrict__ dst, size_t n) {
    size_t i = (size_t)blockIdx.x * blockDim.x + threadIdx.x;
    if (i < n) dst[i] = (__bf16)src[i];
}

// Pack W (K x N, row-major f32) into B-operand layout:
// out[((ntile*KT + ktile)*16 + n)*32 + k] = W[(ktile*32+k)*N + ntile*16+n]
__global__ __launch_bounds__(256) void k_pack_w(const float* __restrict__ W,
                                                __bf16* __restrict__ out, int K, int N) {
    size_t tid = (size_t)blockIdx.x * blockDim.x + threadIdx.x;
    size_t total = (size_t)K * N;
    if (tid >= total) return;
    int k  = (int)(tid & 31);
    int n  = (int)((tid >> 5) & 15);
    size_t r  = tid >> 9;
    int KT = K >> 5;
    int kt = (int)(r % KT);
    int nt = (int)(r / KT);
    out[tid] = (__bf16)W[(size_t)(kt * 32 + k) * N + nt * 16 + n];
}

// ---------------- WMMA GEMM: C = relu?(A @ Bpacked + bias) ----------------
// Block: 256 threads = 8 waves -> 128 rows x 64 cols of C.
// Each wave: one 16-row m-tile x 4 n-tiles (A regs reused 4x).
// B panel staged into LDS in 64col x 256K slices (32 KB) with async copies.
template<bool RELU, bool OUTF32>
__global__ __launch_bounds__(256) void k_gemm_wmma(const __bf16* __restrict__ A,
                                                   const __bf16* __restrict__ Bp,
                                                   const float*  __restrict__ bias,
                                                   __bf16* __restrict__ Cb,
                                                   float*  __restrict__ Cf,
                                                   int N, int K) {
    __shared__ __bf16 sB[16384];   // 4 slabs * 8 ktiles * 16 cols * 32 k = 32 KB

    const int tid   = threadIdx.x;
    const int wave  = tid >> 5;
    const int lane  = tid & 31;
    const int half  = lane >> 4;
    const int mr    = lane & 15;
    const int mtile = blockIdx.x * 8 + wave;
    const int nt0   = blockIdx.y * 4;          // first of 4 n-tiles
    const int KT    = K >> 5;
    const int NCH   = KT >> 3;                 // chunks of 8 k-tiles (256 K)

    const __bf16* arow  = A + (size_t)(mtile * 16 + mr) * K + half * 8;
    const unsigned sbase = (unsigned)(uintptr_t)(void*)sB;

    v8f c[4] = {};
    for (int ch = 0; ch < NCH; ++ch) {
        // ---- stage 64col x 256K slice of packed B into LDS (async) ----
        // 2048 x 16B units; slab = i>>9 (512 units each), contiguous per slab.
        for (int i = tid; i < 2048; i += 256) {
            const __bf16* gsrc = Bp + (size_t)(nt0 + (i >> 9)) * K * 16
                                    + (size_t)ch * 4096 + (size_t)(i & 511) * 8;
            async_copy_b128(sbase + (unsigned)i * 16u, (const void*)gsrc);
        }
        wait_async0();
        __syncthreads();

        // ---- compute 8 k-tiles against the staged slice ----
        const __bf16* achunk = arow + ch * 256;
        #pragma unroll
        for (int ktl = 0; ktl < 8; ++ktl) {
            const __bf16* ap = achunk + ktl * 32;
            __builtin_prefetch(ap + 256, 0, 1);
            v16bf av = cat16(load8(ap), load8(ap + 16));
            #pragma unroll
            for (int q = 0; q < 4; ++q) {
                const __bf16* bp = &sB[((q * 8 + ktl) * 16 + mr) * 32 + half * 8];
                v16bf bv = cat16(load8(bp), load8(bp + 16));
                c[q] = __builtin_amdgcn_wmma_f32_16x16x32_bf16(false, av, false, bv,
                                                               (short)0, c[q], false, false);
            }
        }
        __syncthreads();
    }

    // ---- epilogue: bias (+relu), store ----
    #pragma unroll
    for (int q = 0; q < 4; ++q) {
        const int   col  = (nt0 + q) * 16 + mr;
        const float bcol = bias[col];
        #pragma unroll
        for (int j = 0; j < 8; ++j) {
            float v = c[q][j] + bcol;
            if (RELU) v = v > 0.f ? v : 0.f;
            int row = mtile * 16 + half * 8 + j;
            if (OUTF32) Cf[(size_t)row * N + col] = v;
            else        Cb[(size_t)row * N + col] = (__bf16)v;
        }
    }
}

// ---------------- batch-norm: deterministic column stats + in-place apply ----------------

__global__ __launch_bounds__(256) void k_bn_stats(const __bf16* __restrict__ h,
                                                  float* __restrict__ stats,
                                                  int N, int rows) {
    const int col = blockIdx.x;
    float s = 0.f, ss = 0.f;
    for (int r = threadIdx.x; r < rows; r += 256) {
        float v = (float)h[(size_t)r * N + col];
        s += v; ss += v * v;
    }
    __shared__ float sm[256], sq[256];
    sm[threadIdx.x] = s; sq[threadIdx.x] = ss;
    __syncthreads();
    for (int o = 128; o > 0; o >>= 1) {
        if (threadIdx.x < o) {
            sm[threadIdx.x] += sm[threadIdx.x + o];
            sq[threadIdx.x] += sq[threadIdx.x + o];
        }
        __syncthreads();
    }
    if (threadIdx.x == 0) {
        stats[2 * col]     = sm[0];
        stats[2 * col + 1] = sq[0];
    }
}

__global__ __launch_bounds__(256) void k_bn_apply(__bf16* __restrict__ h,
                                                  const float* __restrict__ stats,
                                                  const float* __restrict__ g,
                                                  const float* __restrict__ beta,
                                                  int N, size_t total, float invB) {
    size_t i = (size_t)blockIdx.x * blockDim.x + threadIdx.x;
    if (i >= total) return;
    int col = (int)(i % N);
    float m   = stats[2 * col] * invB;
    float var = stats[2 * col + 1] * invB - m * m;
    float sc  = g[col] * rsqrtf(var + EPS_);
    float v   = (float)h[i];
    h[i] = (__bf16)((v - m) * sc + beta[col]);
}

// ---------------- per-batch gram -> upper-tri pairs (WMMA) ----------------
// One wave per batch; 3 tiles: (0,0),(0,1),(1,1); K = 64 as 2 k-tiles of 32.
__global__ __launch_bounds__(256) void k_gram_pairs(const __bf16* __restrict__ xb,
                                                    float* __restrict__ pairs) {
    const int b    = blockIdx.x * 8 + (threadIdx.x >> 5);
    const int lane = threadIdx.x & 31;
    const int half = lane >> 4;
    const int mr   = lane & 15;
    const __bf16* xbat = xb + (size_t)b * (NFEAT * D_);

    const int TI[3] = {0, 0, 1};
    const int TJ[3] = {0, 1, 1};
    #pragma unroll
    for (int t = 0; t < 3; ++t) {
        const int ti = TI[t], tj = TJ[t];
        v8f c = {};
        #pragma unroll
        for (int kt = 0; kt < 2; ++kt) {
            const __bf16* ap = xbat + (size_t)(ti * 16 + mr) * D_ + kt * 32 + half * 8;
            const __bf16* bp = xbat + (size_t)(tj * 16 + mr) * D_ + kt * 32 + half * 8;
            v16bf av = cat16(load8(ap), load8(ap + 16));
            v16bf bv = cat16(load8(bp), load8(bp + 16));
            c = __builtin_amdgcn_wmma_f32_16x16x32_bf16(false, av, false, bv,
                                                        (short)0, c, false, false);
        }
        #pragma unroll
        for (int j = 0; j < 8; ++j) {
            int i  = ti * 16 + half * 8 + j;
            int jc = tj * 16 + mr;
            if (i < jc) {
                int idx = i * 31 - (i * (i - 1)) / 2 + (jc - i - 1);
                pairs[(size_t)b * NP + idx] = c[j];
            }
        }
    }
}

// ---------------- final: out = [hof | pairs] @ Wc + bc ----------------

__global__ __launch_bounds__(256) void k_final(const float* __restrict__ hof,
                                               const float* __restrict__ pairs,
                                               const float* __restrict__ Wc,
                                               const float* __restrict__ bc,
                                               float* __restrict__ out) {
    int b = blockIdx.x * blockDim.x + threadIdx.x;
    if (b >= B_) return;
    float acc = bc[0];
    const float* h = hof + (size_t)b * HOF_;
    #pragma unroll 8
    for (int k = 0; k < HOF_; ++k) acc += h[k] * Wc[k];
    const float* p = pairs + (size_t)b * NP;
    for (int k = 0; k < NP; ++k) acc += p[k] * Wc[HOF_ + k];
    out[b] = acc;
}

// ---------------- host launch ----------------

extern "C" void kernel_launch(void* const* d_in, const int* in_sizes, int n_in,
                              void* d_out, int out_size, void* d_ws, size_t ws_size,
                              hipStream_t stream) {
    const float* x   = (const float*)d_in[0];
    const float* W1  = (const float*)d_in[1];
    const float* b1  = (const float*)d_in[2];
    const float* g1  = (const float*)d_in[3];
    const float* be1 = (const float*)d_in[4];
    const float* W2  = (const float*)d_in[5];
    const float* b2  = (const float*)d_in[6];
    const float* g2  = (const float*)d_in[7];
    const float* be2 = (const float*)d_in[8];
    const float* W3  = (const float*)d_in[9];
    const float* b3  = (const float*)d_in[10];
    const float* Wc  = (const float*)d_in[11];
    const float* bc  = (const float*)d_in[12];
    float* out = (float*)d_out;

    char* w = (char*)d_ws;
    auto alloc = [&](size_t bytes) -> char* {
        char* p = w; w += (bytes + 255) & ~(size_t)255; return p;
    };
    __bf16* xb   = (__bf16*)alloc((size_t)B_ * K1 * 2);
    __bf16* W1p  = (__bf16*)alloc((size_t)K1 * H1 * 2);
    __bf16* W2p  = (__bf16*)alloc((size_t)H1 * H2 * 2);
    __bf16* W3p  = (__bf16*)alloc((size_t)H2 * HOF_ * 2);
    __bf16* h1   = (__bf16*)alloc((size_t)B_ * H1 * 2);
    __bf16* h2   = (__bf16*)alloc((size_t)B_ * H2 * 2);
    float*  hof  = (float*) alloc((size_t)B_ * HOF_ * 4);
    float*  prs  = (float*) alloc((size_t)B_ * NP * 4);
    float*  st1  = (float*) alloc((size_t)2 * H1 * 4);
    float*  st2  = (float*) alloc((size_t)2 * H2 * 4);

    // 1. x -> bf16
    {
        size_t n = (size_t)B_ * K1;
        k_cvt_bf16<<<dim3((unsigned)((n + 255) / 256)), dim3(256), 0, stream>>>(x, xb, n);
    }
    // 2. pack weights
    k_pack_w<<<dim3((K1 * H1) / 256),   dim3(256), 0, stream>>>(W1, W1p, K1, H1);
    k_pack_w<<<dim3((H1 * H2) / 256),   dim3(256), 0, stream>>>(W2, W2p, H1, H2);
    k_pack_w<<<dim3((H2 * HOF_) / 256), dim3(256), 0, stream>>>(W3, W3p, H2, HOF_);
    // 3. gram pairs (needs xb only)
    k_gram_pairs<<<dim3(B_ / 8), dim3(256), 0, stream>>>(xb, prs);
    // 4. fc1 + relu
    k_gemm_wmma<true, false><<<dim3(B_ / 128, H1 / 64), dim3(256), 0, stream>>>(
        xb, W1p, b1, h1, nullptr, H1, K1);
    // 5. bn1
    k_bn_stats<<<dim3(H1), dim3(256), 0, stream>>>(h1, st1, H1, B_);
    {
        size_t n = (size_t)B_ * H1;
        k_bn_apply<<<dim3((unsigned)((n + 255) / 256)), dim3(256), 0, stream>>>(
            h1, st1, g1, be1, H1, n, 1.0f / B_);
    }
    // 6. fc2 + relu
    k_gemm_wmma<true, false><<<dim3(B_ / 128, H2 / 64), dim3(256), 0, stream>>>(
        h1, W2p, b2, h2, nullptr, H2, H1);
    // 7. bn2
    k_bn_stats<<<dim3(H2), dim3(256), 0, stream>>>(h2, st2, H2, B_);
    {
        size_t n = (size_t)B_ * H2;
        k_bn_apply<<<dim3((unsigned)((n + 255) / 256)), dim3(256), 0, stream>>>(
            h2, st2, g2, be2, H2, n, 1.0f / B_);
    }
    // 8. fc3 (f32 out, no relu)
    k_gemm_wmma<false, true><<<dim3(B_ / 128, HOF_ / 64), dim3(256), 0, stream>>>(
        h2, W3p, b3, nullptr, hof, HOF_, H2);
    // 9. combine
    k_final<<<dim3(B_ / 256), dim3(256), 0, stream>>>(hof, prs, Wc, bc, out);
}
